// Head_10505490006021
// MI455X (gfx1250) — matrix operand order repaired and checked by reading, not compiled, not run on hardware
//
#include <hip/hip_runtime.h>
#include <hip/hip_bf16.h>

#define N_EMBD 1024
#define HEAD   128
#define BATCH  8
#define SEQ    4096

#define KLDS_STRIDE 136   // 128 + 8 pad: 16 fragment lanes hit 16 banks
#define VLDS_STRIDE 40    // 32 + 8 pad

typedef __attribute__((ext_vector_type(16))) __bf16 v16bf;
typedef __attribute__((ext_vector_type(8)))  __bf16 v8bf;
typedef __attribute__((ext_vector_type(8)))  float  v8f;

static __device__ __forceinline__ v8f wmma_bf16(v16bf a, v16bf b, v8f c) {
    // D = A(16x32) * B(32x16) + C, f32 accumulate
    return __builtin_amdgcn_wmma_f32_16x16x32_bf16(
        /*neg_a=*/false, a, /*neg_b=*/false, b,
        /*c_mod=*/(short)0, c, /*reuse_a=*/false, /*reuse_b=*/false);
}

// ---------------------------------------------------------------------------
// Kernel 0: Wt[w][n][k] = (bf16)W_w[k][n]  -- W transposed+converted, once.
// ---------------------------------------------------------------------------
__global__ __launch_bounds__(256)
void wtrans_kernel(const float* __restrict__ Wq, const float* __restrict__ Wk,
                   const float* __restrict__ Wv, __bf16* __restrict__ wt)
{
    const int idx = blockIdx.x * 256 + threadIdx.x;   // 0 .. 3*HEAD*N_EMBD-1
    const int w   = idx / (HEAD * N_EMBD);
    const int rem = idx % (HEAD * N_EMBD);
    const int n   = rem / N_EMBD;
    const int k   = rem % N_EMBD;
    const float* W = (w == 0) ? Wq : (w == 1) ? Wk : Wv;
    wt[idx] = (__bf16)W[(size_t)k * HEAD + n];
}

// ---------------------------------------------------------------------------
// Kernel 1: q/k/v = x @ W + b, bf16 out. One wave per 16-row tile.
// blockIdx.y selects q/k/v.  v is stored TRANSPOSED per batch: Vt[b][h][t].
// ---------------------------------------------------------------------------
__global__ __launch_bounds__(32)
void qkv_proj_kernel(const float* __restrict__ x,
                     const __bf16* __restrict__ wt,   // [3][HEAD][N_EMBD]
                     const float* __restrict__ bq, const float* __restrict__ bk,
                     const float* __restrict__ bv,
                     __bf16* __restrict__ qo, __bf16* __restrict__ ko,
                     __bf16* __restrict__ vt)
{
    const int lane = threadIdx.x;
    const int m  = lane & 15;
    const int hi = lane >> 4;
    const int which = blockIdx.y;

    const __bf16* W = wt + (size_t)which * HEAD * N_EMBD;
    const float* bias = (which == 0) ? bq : (which == 1) ? bk : bv;

    const int r0 = blockIdx.x * 16;                    // first row (over B*T)
    const float* xrow = x + (size_t)(r0 + m) * N_EMBD; // A: lane = row m

    v8f acc[8];
#pragma unroll
    for (int c = 0; c < 8; ++c) acc[c] = (v8f){0,0,0,0,0,0,0,0};

    for (int k0 = 0; k0 < N_EMBD; k0 += 32) {
        v16bf a;
#pragma unroll
        for (int i = 0; i < 8; ++i) {
            a[i]     = (__bf16)xrow[k0 + hi * 16 + i];
            a[i + 8] = (__bf16)xrow[k0 + hi * 16 + 8 + i];
        }
#pragma unroll
        for (int c = 0; c < 8; ++c) {
            v16bf bfr = *(const v16bf*)(W + (size_t)(c * 16 + m) * N_EMBD
                                          + k0 + hi * 16);
            acc[c] = wmma_bf16(a, bfr, acc[c]);
        }
    }

    if (which < 2) {
        __bf16* out = (which == 0) ? qo : ko;
#pragma unroll
        for (int c = 0; c < 8; ++c) {
            const float bb = bias[c * 16 + m];
#pragma unroll
            for (int r = 0; r < 8; ++r)
                out[(size_t)(r0 + r + hi * 8) * HEAD + c * 16 + m] =
                    (__bf16)(acc[c][r] + bb);
        }
    } else {
        const int b  = r0 >> 12;          // / SEQ
        const int t0 = r0 & (SEQ - 1);
        __bf16* vtb = vt + (size_t)b * HEAD * SEQ;
#pragma unroll
        for (int c = 0; c < 8; ++c) {
            const float bb = bias[c * 16 + m];
#pragma unroll
            for (int r = 0; r < 8; ++r)
                vtb[(size_t)(c * 16 + m) * SEQ + t0 + r + hi * 8] =
                    (__bf16)(acc[c][r] + bb);
        }
    }
}

// ---------------------------------------------------------------------------
// Kernel 2: causal flash attention, transposed formulation, 4 waves / WG.
// The workgroup covers 64 query rows; each KV tile (K: 32x128, Vt: 128x32)
// is loaded cooperatively into LDS once and consumed by all 4 waves,
// cutting L2 read traffic 4x.
// ---------------------------------------------------------------------------
__global__ __launch_bounds__(128)
void attn_kernel(const __bf16* __restrict__ Q, const __bf16* __restrict__ K,
                 const __bf16* __restrict__ Vt, float* __restrict__ out)
{
    __shared__ __bf16 Klds[32 * KLDS_STRIDE];       // K[kv_local][h]
    __shared__ __bf16 Vlds[128 * VLDS_STRIDE];      // Vt[h][kv_local]
    __shared__ __bf16 Plds[4][16 * 32];             // per-wave P[q][kv_local]

    const int tid  = threadIdx.x;
    const int wid  = tid >> 5;
    const int lane = tid & 31;
    const int m  = lane & 15;
    const int hi = lane >> 4;

    const int blk   = blockIdx.x;            // 0 .. B*T/64-1
    const int b     = blk >> 6;              // / (SEQ/64)
    const int qbase = (blk & 63) << 6;
    const int q0    = qbase + wid * 16;      // this wave's query tile
    const size_t base  = (size_t)b * SEQ * HEAD;
    const __bf16* vtb  = Vt + (size_t)b * HEAD * SEQ;

    // Preload Q as 4 B-fragments: B[h][n] = Q[q0+n][h0+h]; lane n=m,
    // elements h = hi*16+i -> one contiguous 32B load each.
    const __bf16* qrow = Q + base + (size_t)(q0 + m) * HEAD + hi * 16;
    v16bf qb[4];
#pragma unroll
    for (int t = 0; t < 4; ++t)
        qb[t] = *(const v16bf*)(qrow + t * 32);

    v8f o[8];                    // O^T chunks: o[c][r] = out[q0+m][c*16+r+hi*8]
#pragma unroll
    for (int c = 0; c < 8; ++c) o[c] = (v8f){0,0,0,0,0,0,0,0};
    float mval = -1e30f, lval = 0.0f;   // per-lane (q = q0+m) softmax stats

    const float scale = 0.08838834764831845f;  // 1/sqrt(128)

    // thread's slice of the cooperative loads
    const int krow = tid >> 2;               // 0..31
    const int kchk = (tid & 3) * 32;         // bf16 offset within K row

    for (int kv0 = 0; kv0 < qbase + 64; kv0 += 32) {
        // ---- cooperative K/V tile load into LDS (8KB + 8KB) ----
        {
            const __bf16* gk = K + base + (size_t)(kv0 + krow) * HEAD + kchk;
            __bf16* lk = Klds + krow * KLDS_STRIDE + kchk;
#pragma unroll
            for (int j = 0; j < 4; ++j)
                *(uint4*)(lk + j * 8) = *(const uint4*)(gk + j * 8);

            const __bf16* gv = vtb + (size_t)tid * SEQ + kv0;   // h row = tid
            __bf16* lv = Vlds + tid * VLDS_STRIDE;
#pragma unroll
            for (int j = 0; j < 4; ++j)
                *(uint4*)(lv + j * 8) = *(const uint4*)(gv + j * 8);
        }
        __syncthreads();

        if (kv0 < q0 + 16) {     // wave-uniform: this wave still has work
            // ---- S^T = K * Q^T : two 16-kv sub-tiles, A from LDS ----
            v8f s0 = (v8f){0,0,0,0,0,0,0,0};
            v8f s1 = (v8f){0,0,0,0,0,0,0,0};
            const __bf16* k0r = Klds + m * KLDS_STRIDE;
            const __bf16* k1r = k0r + 16 * KLDS_STRIDE;
#pragma unroll
            for (int t = 0; t < 4; ++t) {
                const int h0 = t * 32;
                v16bf ka;
#pragma unroll
                for (int i = 0; i < 8; ++i) {
                    ka[i]     = k0r[h0 + hi * 16 + i];
                    ka[i + 8] = k0r[h0 + hi * 16 + 8 + i];
                }
                s0 = wmma_bf16(ka, qb[t], s0);
            }
#pragma unroll
            for (int t = 0; t < 4; ++t) {
                const int h0 = t * 32;
                v16bf ka;
#pragma unroll
                for (int i = 0; i < 8; ++i) {
                    ka[i]     = k1r[h0 + hi * 16 + i];
                    ka[i + 8] = k1r[h0 + hi * 16 + 8 + i];
                }
                s1 = wmma_bf16(ka, qb[t], s1);
            }

            // ---- online softmax: per-lane kv values, q = q0+m fixed ----
            float xs[16];
            if (kv0 + 31 <= q0) {          // fully unmasked (uniform branch)
#pragma unroll
                for (int r = 0; r < 8; ++r) {
                    xs[r]     = s0[r] * scale;
                    xs[r + 8] = s1[r] * scale;
                }
            } else {
                const int q = q0 + m;
#pragma unroll
                for (int r = 0; r < 8; ++r) {
                    const int kva = kv0 + r + hi * 8;
                    xs[r]     = (kva      <= q) ? s0[r] * scale : -1e30f;
                    xs[r + 8] = (kva + 16 <= q) ? s1[r] * scale : -1e30f;
                }
            }
            float tmax = xs[0];
#pragma unroll
            for (int i = 1; i < 16; ++i) tmax = fmaxf(tmax, xs[i]);
            tmax = fmaxf(tmax, __shfl_xor(tmax, 16));
            const float mnew  = fmaxf(mval, tmax);
            const float alpha = __expf(mval - mnew);
            mval = mnew;

            v8bf pl0, pl1;
            float rs = 0.0f;
#pragma unroll
            for (int r = 0; r < 8; ++r) {
                const float p = __expf(xs[r] - mnew);
                rs += p;  pl0[r] = (__bf16)p;
            }
#pragma unroll
            for (int r = 0; r < 8; ++r) {
                const float p = __expf(xs[r + 8] - mnew);
                rs += p;  pl1[r] = (__bf16)p;
            }
            rs += __shfl_xor(rs, 16);
            lval = lval * alpha + rs;
#pragma unroll
            for (int c = 0; c < 8; ++c)
#pragma unroll
                for (int r = 0; r < 8; ++r) o[c][r] *= alpha;

            // P[q=m][kv] to this wave's LDS slice (same-wave DS is in-order)
            __bf16* pw = Plds[wid];
            *(v8bf*)(pw + m * 32 + hi * 8)      = pl0;
            *(v8bf*)(pw + m * 32 + 16 + hi * 8) = pl1;

            // P^T B-fragment: lane n=m, kv = hi*16+i -> contiguous 32B
            const v16bf pb = *(const v16bf*)(pw + m * 32 + hi * 16);

            // ---- O^T += V^T * P^T, A fragments from LDS ----
#pragma unroll
            for (int c = 0; c < 8; ++c) {
                const __bf16* vr = Vlds + (c * 16 + m) * VLDS_STRIDE;
                v16bf va;
#pragma unroll
                for (int i = 0; i < 8; ++i) {
                    va[i]     = vr[hi * 16 + i];
                    va[i + 8] = vr[hi * 16 + 8 + i];
                }
                o[c] = wmma_bf16(va, pb, o[c]);
            }
        }
        __syncthreads();   // protect K/V LDS tiles before next overwrite
    }

    // ---- epilogue: O^T / l -> row-major out, contiguous v8f stores ----
    const float inv = 1.0f / lval;
    float* orow = out + base + (size_t)(q0 + m) * HEAD + hi * 8;
#pragma unroll
    for (int c = 0; c < 8; ++c) {
        v8f ov;
#pragma unroll
        for (int r = 0; r < 8; ++r) ov[r] = o[c][r] * inv;
        *(v8f*)(orow + c * 16) = ov;
    }
}

extern "C" void kernel_launch(void* const* d_in, const int* in_sizes, int n_in,
                              void* d_out, int out_size, void* d_ws, size_t ws_size,
                              hipStream_t stream) {
    (void)in_sizes; (void)n_in; (void)out_size; (void)ws_size;
    const float* x  = (const float*)d_in[0];
    const float* Wq = (const float*)d_in[1];
    const float* bq = (const float*)d_in[2];
    const float* Wk = (const float*)d_in[3];
    const float* bk = (const float*)d_in[4];
    const float* Wv = (const float*)d_in[5];
    const float* bv = (const float*)d_in[6];
    float* out = (float*)d_out;

    const size_t per = (size_t)BATCH * SEQ * HEAD;   // elements per matrix
    __bf16* qw = (__bf16*)d_ws;                       // 8 MB
    __bf16* kw = qw + per;                            // 8 MB
    __bf16* vt = kw + per;                            // 8 MB (transposed)
    __bf16* wt = vt + per;                            // 0.75 MB

    wtrans_kernel<<<3 * HEAD * N_EMBD / 256, 256, 0, stream>>>(Wq, Wk, Wv, wt);

    dim3 pgrid(BATCH * SEQ / 16, 3);
    qkv_proj_kernel<<<pgrid, 32, 0, stream>>>(x, wt, bq, bk, bv, qw, kw, vt);

    attn_kernel<<<BATCH * SEQ / 64, 128, 0, stream>>>(qw, kw, vt, out);
}
